// HybridANFIS_38534446580295
// MI455X (gfx1250) — matrix operand order repaired and checked by reading, not compiled, not run on hardware
//
#include <hip/hip_runtime.h>
#include <hip/hip_fp16.h>

typedef __attribute__((ext_vector_type(16))) _Float16 v16h;
typedef __attribute__((ext_vector_type(8)))  float    v8f;

#define B_N   8192
#define D_N   16
#define M_N   3
#define R_N   1000
#define C_N   10
#define KPAD  64
#define NPAD  1008
#define NT    (NPAD / 16)      // 63 N-tiles
#define TOPK  200
#define WAVES 8
#define LDS_STRIDE 1012        // bank-friendly row stride for the LDS firing tile

// workspace byte offsets
#define OFF_A      0                                   // B_N*KPAD*2      = 1,048,576
#define OFF_B      (B_N * KPAD * 2)                    // KPAD*NPAD*2     =   129,024
#define OFF_SCONS  (OFF_B + KPAD * NPAD * 2)           // R_N*C_N*4       =    40,000
#define OFF_SX     (OFF_SCONS + R_N * C_N * 4)         // B_N*4           =    32,768

// ---------------------------------------------------------------- kernel 1
__global__ __launch_bounds__(256)
void anfis_prep_a(const float* __restrict__ x,
                  const float* __restrict__ centers,
                  const float* __restrict__ widths,
                  _Float16* __restrict__ Ah,
                  float* __restrict__ x_ext,
                  float* __restrict__ s_x)
{
    int b = blockIdx.x * blockDim.x + threadIdx.x;
    if (b >= B_N) return;
    float xs[D_N];
    float s = 0.f;
    #pragma unroll
    for (int d = 0; d < D_N; ++d) {
        float v = x[b * D_N + d];
        xs[d] = v; s += v;
        x_ext[b * (D_N + 1) + d] = v;
    }
    x_ext[b * (D_N + 1) + D_N] = 1.f;
    s_x[b] = s + 1.f;
    #pragma unroll
    for (int d = 0; d < D_N; ++d) {
        #pragma unroll
        for (int m = 0; m < M_N; ++m) {
            float c = centers[d * M_N + m];
            float w = widths[d * M_N + m];
            float df = xs[d] - c;
            float lmf = -(df * df) / (2.f * w * w);
            Ah[(size_t)b * KPAD + d * M_N + m] = (_Float16)lmf;
        }
    }
    #pragma unroll
    for (int kk = D_N * M_N; kk < KPAD; ++kk)
        Ah[(size_t)b * KPAD + kk] = (_Float16)0.f;
}

// ---------------------------------------------------------------- kernel 2
__global__ __launch_bounds__(256)
void anfis_pack_b(const int* __restrict__ rules, _Float16* __restrict__ Bh)
{
    int r = blockIdx.x * blockDim.x + threadIdx.x;
    if (r >= NPAD) return;
    if (r < R_N) {
        for (int d = 0; d < D_N; ++d) {
            int mm = rules[r * D_N + d];
            #pragma unroll
            for (int m = 0; m < M_N; ++m)
                Bh[(size_t)(d * M_N + m) * NPAD + r] = (_Float16)((mm == m) ? 1.f : 0.f);
        }
    } else {
        for (int kk = 0; kk < D_N * M_N; ++kk)
            Bh[(size_t)kk * NPAD + r] = (_Float16)0.f;
    }
    #pragma unroll
    for (int kk = D_N * M_N; kk < KPAD; ++kk)
        Bh[(size_t)kk * NPAD + r] = (_Float16)0.f;
}

__global__ __launch_bounds__(256)
void anfis_scons(const float* __restrict__ cons, float* __restrict__ scons)
{
    int t = blockIdx.x * blockDim.x + threadIdx.x;
    if (t >= R_N * C_N) return;
    int r = t / C_N, c = t % C_N;
    float s = 0.f;
    #pragma unroll
    for (int j = 0; j < D_N + 1; ++j)
        s += cons[((size_t)r * (D_N + 1) + j) * C_N + c];
    scons[t] = s;
}

// ---------------------------------------------------------------- fused kernel
// One workgroup = 16 batch rows. Phase 1: 8 waves sweep the 63 rule-tiles with
// v_wmma_f32_16x16x32_f16, exp() into an LDS firing tile. Phase 2: each wave
// runs exact radix-select top-200 + normalize + s_cons dot for 2 rows, writing
// norm_firing to HBM exactly once.
__global__ __launch_bounds__(256)
void anfis_fused(const _Float16* __restrict__ Ah,
                 const _Float16* __restrict__ Bh,
                 const float* __restrict__ scons,
                 const float* __restrict__ s_x,
                 float* __restrict__ norm,
                 float* __restrict__ y_hat)
{
    __shared__ float    s_fire[16 * LDS_STRIDE];       // 64,768 B
    __shared__ unsigned s_hist[WAVES * 256];           //  8,192 B

    const int tid  = threadIdx.x;
    const int lane = tid & 31;
    const int w    = tid >> 5;            // wave id 0..7
    const int b0   = blockIdx.x * 16;
    const int row  = lane & 15;
    const int hl   = lane >> 4;

    union F16x16 { v16h h; uint4 q[2]; };

    // ---- Phase 1: A fragment (reused across all tiles of this wave) ----
    F16x16 fa[2];
    #pragma unroll
    for (int qk = 0; qk < 2; ++qk) {
        const size_t abase = (size_t)(b0 + row) * KPAD + qk * 32;
        fa[qk].q[0] = *(const uint4*)(Ah + abase + hl * 8);
        fa[qk].q[1] = *(const uint4*)(Ah + abase + 16 + hl * 8);
    }

    for (int nt = w; nt < NT; nt += WAVES) {
        const int n0 = nt * 16;
        v8f acc = {};
        #pragma unroll
        for (int qk = 0; qk < 2; ++qk) {
            F16x16 fb;
            #pragma unroll
            for (int s = 0; s < 16; ++s) {
                const int kk = qk * 32 + ((s < 8) ? (hl * 8 + s) : (16 + hl * 8 + (s - 8)));
                fb.h[s] = Bh[(size_t)kk * NPAD + n0 + row];
            }
            acc = __builtin_amdgcn_wmma_f32_16x16x32_f16(
                false, fa[qk].h, false, fb.h, (short)0, acc, false, false);
        }
        // D layout: lane<16 -> N=lane, VGPR i -> M=i ; lane>=16 -> N=lane-16, M=8+i
        #pragma unroll
        for (int i = 0; i < 8; ++i) {
            int m = (hl == 0) ? i : (8 + i);
            s_fire[m * LDS_STRIDE + n0 + row] = __expf(acc[i]);
        }
    }
    __syncthreads();

    // ---- Phase 2: top-k + normalize + y_hat, 2 rows per wave ----
    unsigned* hist = s_hist + w * 256;
    for (int rr = 0; rr < 2; ++rr) {
        const int myrow = w * 2 + rr;                  // 0..15
        float* frow = s_fire + myrow * LDS_STRIDE;

        unsigned pref = 0u, remk = TOPK;
        for (int p = 3; p >= 0; --p) {
            #pragma unroll
            for (int i = 0; i < 8; ++i) hist[lane * 8 + i] = 0u;
            __syncthreads();
            const unsigned himask = (p == 3) ? 0u : (0xFFFFFFFFu << ((p + 1) * 8));
            for (int r = lane; r < R_N; r += 32) {
                unsigned u = __float_as_uint(frow[r]);
                if ((u & himask) == pref)
                    atomicAdd(&hist[(u >> (p * 8)) & 255u], 1u);
            }
            __syncthreads();
            unsigned dg = 0u, nrem = remk;
            if (lane == 0) {
                unsigned cum = 0u;
                for (int d = 255; d >= 0; --d) {
                    unsigned c = hist[d];
                    if (cum + c >= remk) { dg = (unsigned)d; nrem = remk - cum; break; }
                    cum += c;
                }
            }
            pref = __shfl(pref | (dg << (p * 8)), 0);
            remk = __shfl(nrem, 0);
            __syncthreads();
        }
        const unsigned T = pref;

        // ties at T: keep lowest indices first (matches top_k semantics)
        if (lane == 0) {
            unsigned budget = remk;
            for (int r = 0; r < R_N; ++r) {
                if (__float_as_uint(frow[r]) == T) {
                    if (budget > 0) budget--;
                    else            frow[r] = 0.f;
                }
            }
        }
        __syncthreads();

        // mask, masked sum, s_cons dot products
        float acc[C_N];
        #pragma unroll
        for (int c = 0; c < C_N; ++c) acc[c] = 0.f;
        float psum = 0.f;
        for (int r = lane; r < R_N; r += 32) {
            float v = frow[r];
            unsigned u = __float_as_uint(v);
            v = (u >= T) ? v : 0.f;
            frow[r] = v;
            psum += v;
            #pragma unroll
            for (int c = 0; c < C_N; ++c)
                acc[c] += v * scons[r * C_N + c];
        }
        #pragma unroll
        for (int off = 16; off > 0; off >>= 1) {
            psum += __shfl_down(psum, off);
            #pragma unroll
            for (int c = 0; c < C_N; ++c) acc[c] += __shfl_down(acc[c], off);
        }
        const float total = __shfl(psum, 0);
        const float inv = 1.f / (total + 1e-9f);

        float* grow = norm + (size_t)(b0 + myrow) * R_N;
        for (int r = lane; r < R_N; r += 32)
            grow[r] = frow[r] * inv;
        if (lane == 0) {
            const float sx = s_x[b0 + myrow];
            #pragma unroll
            for (int c = 0; c < C_N; ++c)
                y_hat[(size_t)(b0 + myrow) * C_N + c] = sx * inv * acc[c];
        }
        __syncthreads();
    }
}

// ---------------------------------------------------------------- launch
extern "C" void kernel_launch(void* const* d_in, const int* in_sizes, int n_in,
                              void* d_out, int out_size, void* d_ws, size_t ws_size,
                              hipStream_t stream)
{
    (void)in_sizes; (void)n_in; (void)out_size; (void)ws_size;
    const float* x       = (const float*)d_in[0];
    const float* centers = (const float*)d_in[1];
    const float* widths  = (const float*)d_in[2];
    const int*   rules   = (const int*)d_in[3];
    const float* cons    = (const float*)d_in[4];

    float* out   = (float*)d_out;
    float* y_hat = out;                                   // [B, C]
    float* norm  = out + (size_t)B_N * C_N;               // [B, R]
    float* x_ext = norm + (size_t)B_N * R_N;              // [B, D+1]

    char* ws = (char*)d_ws;
    _Float16* Ah    = (_Float16*)(ws + OFF_A);
    _Float16* Bh    = (_Float16*)(ws + OFF_B);
    float*    scons = (float*)(ws + OFF_SCONS);
    float*    sx    = (float*)(ws + OFF_SX);

    anfis_prep_a<<<B_N / 256, 256, 0, stream>>>(x, centers, widths, Ah, x_ext, sx);
    anfis_pack_b<<<(NPAD + 255) / 256, 256, 0, stream>>>(rules, Bh);
    anfis_scons<<<(R_N * C_N + 255) / 256, 256, 0, stream>>>(cons, scons);

    anfis_fused<<<B_N / 16, 256, 0, stream>>>(Ah, Bh, scons, sx, norm, y_hat);
}
